// MultiHeadAttention_24240795419279
// MI455X (gfx1250) — compile-verified, matrix operand order checked by reading, not run on hardware
//
#include <hip/hip_runtime.h>
#include <hip/hip_bf16.h>
#include <stdint.h>

#define S_LEN   4096
#define DMODEL  2048
#define KVDIM   512
#define HEADS   16
#define KVHEADS 4
#define HDIM    128

typedef __bf16 bf16x16 __attribute__((ext_vector_type(16)));
typedef float  f32x8   __attribute__((ext_vector_type(8)));
typedef int    v4i     __attribute__((ext_vector_type(4)));

#define GLOBAL_AS __attribute__((address_space(1)))
#define LDS_AS    __attribute__((address_space(3)))

#if __has_builtin(__builtin_amdgcn_global_load_async_to_lds_b128)
#define HAVE_ASYNC_LDS 1
#else
#define HAVE_ASYNC_LDS 0
#endif

__device__ __forceinline__ uint16_t f2bf(float f) {
    uint32_t u = __float_as_uint(f);
    u += 0x7fffu + ((u >> 16) & 1u);   // round-to-nearest-even
    return (uint16_t)(u >> 16);
}

// 16B global -> LDS copy: async (ASYNCcnt, bypasses VGPRs) when available,
// else plain load+store (compiler inserts loadcnt/dscnt waits).
__device__ __forceinline__ void copy16_g2l(const uint16_t* g, uint16_t* l) {
#if HAVE_ASYNC_LDS
    __builtin_amdgcn_global_load_async_to_lds_b128(
        (GLOBAL_AS v4i*)g, (LDS_AS v4i*)l, 0, 0);
#else
    *(uint4*)l = *(const uint4*)g;
#endif
}

template <int N>
__device__ __forceinline__ void wait_async() {
#if HAVE_ASYNC_LDS
#if __has_builtin(__builtin_amdgcn_s_wait_asynccnt)
    __builtin_amdgcn_s_wait_asynccnt(N);
#else
    asm volatile("s_wait_asynccnt %0" :: "i"(N) : "memory");
#endif
#endif
}

// Load a 16x32 bf16 A-style fragment (also valid for B loaded from row-major
// [N][K] storage).  Per ISA 16-bit A layout: lane holds row (lane&15); VGPR
// v0..v3 hold k = grp*8 .. grp*8+7, v4..v7 hold k = 16+grp*8 .. 16+grp*8+7.
// Caller passes p = row_base + grp*8; we fetch p[0..7] and p[16..23].
__device__ __forceinline__ bf16x16 load_frag(const uint16_t* p) {
    union { bf16x16 f; uint4 q[2]; } u;
    u.q[0] = *(const uint4*)(p);
    u.q[1] = *(const uint4*)(p + 16);
    return u.f;
}

__device__ __forceinline__ f32x8 wmma_bf16(bf16x16 a, bf16x16 b, f32x8 c) {
    return __builtin_amdgcn_wmma_f32_16x16x32_bf16(
        false, a, false, b, (short)0, c, false, false);
}

// ---------------------------------------------------------------- converts
__global__ void f32_to_bf16_kern(const float* __restrict__ in,
                                 uint16_t* __restrict__ out, int n) {
    int i = blockIdx.x * blockDim.x + threadIdx.x;
    if (i < n) out[i] = f2bf(in[i]);
}

// ---------------------------------------------------------------- GEMM
// C[M,N] (f32) = A[M,K] (bf16, row-major) * B[N,K]^T (bf16, row-major)
// Block tile 256x128, K-step 32, double-buffered LDS fed by async-to-LDS.
// 8 waves: wave (wm in 0..3, wn in 0..1) computes 64x64 (16 WMMAs / k-step).
#define GBM 256
#define GBN 128
#define GBK 32

__global__ __launch_bounds__(256, 1)
void gemm_bf16_nt(const uint16_t* __restrict__ A, const uint16_t* __restrict__ B,
                  float* __restrict__ C, int M, int N, int K) {
    __shared__ __align__(16) uint16_t sA[2][GBM * GBK];   // 2 x 16 KB
    __shared__ __align__(16) uint16_t sB[2][GBN * GBK];   // 2 x  8 KB

    const int tid  = threadIdx.x;
    const int lane = tid & 31;
    const int w    = tid >> 5;
    const int lrow = lane & 15;
    const int grp  = lane >> 4;
    const int m0   = blockIdx.y * GBM;
    const int n0   = blockIdx.x * GBN;
    const int wm   = w >> 1;      // 0..3
    const int wn   = w & 1;       // 0..1

    f32x8 acc[4][4] = {};

    // 6 async 16B copies per thread per tile (4 for A: 1024 chunks, 2 for B: 512)
    auto issue_tile = [&](int k0, int buf) {
        #pragma unroll
        for (int i = 0; i < 4; ++i) {
            int c = tid + i * 256;
            int row = c >> 2, col = (c & 3) * 8;
            copy16_g2l(A + (size_t)(m0 + row) * K + k0 + col,
                       sA[buf] + row * GBK + col);
        }
        #pragma unroll
        for (int i = 0; i < 2; ++i) {
            int c = tid + i * 256;
            int row = c >> 2, col = (c & 3) * 8;
            copy16_g2l(B + (size_t)(n0 + row) * K + k0 + col,
                       sB[buf] + row * GBK + col);
        }
    };

    issue_tile(0, 0);
    const int nk = K / GBK;

    for (int it = 0; it < nk; ++it) {
        const int buf = it & 1;
        if (it + 1 < nk) {
            issue_tile((it + 1) * GBK, buf ^ 1);
            wait_async<6>();    // tile(it) resident; tile(it+1) in flight
        } else {
            wait_async<0>();
        }
        __syncthreads();

        bf16x16 bfr[4];
        #pragma unroll
        for (int nt = 0; nt < 4; ++nt)
            bfr[nt] = load_frag(sB[buf] + (wn * 64 + nt * 16 + lrow) * GBK + grp * 8);
        #pragma unroll
        for (int mt = 0; mt < 4; ++mt) {
            bf16x16 af = load_frag(sA[buf] + (wm * 64 + mt * 16 + lrow) * GBK + grp * 8);
            #pragma unroll
            for (int nt = 0; nt < 4; ++nt)
                acc[mt][nt] = wmma_bf16(af, bfr[nt], acc[mt][nt]);
        }
        __syncthreads();   // all waves done with buf before it is re-filled
    }

    // C/D layout: lanes 0-15 -> rows r, lanes 16-31 -> rows r+8; col = lane&15
    #pragma unroll
    for (int mt = 0; mt < 4; ++mt)
        #pragma unroll
        for (int nt = 0; nt < 4; ++nt)
            #pragma unroll
            for (int r = 0; r < 8; ++r)
                C[(size_t)(m0 + wm * 64 + mt * 16 + r + grp * 8) * N +
                  (n0 + wn * 64 + nt * 16 + lrow)] = acc[mt][nt][r];
}

// ---------------------------------------------------------------- RoPE / pack
// Q f32 [S][2048] -> bf16 [H][S][128], rotary applied, 1/sqrt(128) folded in.
__global__ void rope_pack_q(const float* __restrict__ Q, uint16_t* __restrict__ Qb) {
    int t = blockIdx.x * blockDim.x + threadIdx.x;
    if (t >= S_LEN * HEADS * (HDIM / 2)) return;
    int j = t & 63;
    int h = (t >> 6) & (HEADS - 1);
    int s = t >> 10;
    float inv = __powf(10000.0f, -(float)(2 * j) * (1.0f / 128.0f));
    float sn, cs;
    __sincosf((float)s * inv, &sn, &cs);
    const float* src = Q + (size_t)s * DMODEL + h * HDIM + 2 * j;
    float xr = src[0], xi = src[1];
    const float scale = 0.08838834764831845f;  // 1/sqrt(128)
    uint16_t* dst = Qb + ((size_t)h * S_LEN + s) * HDIM + 2 * j;
    dst[0] = f2bf((xr * cs - xi * sn) * scale);
    dst[1] = f2bf((xr * sn + xi * cs) * scale);
}

// K f32 [S][512] -> bf16 [Hkv][S][128], rotary applied.
__global__ void rope_pack_k(const float* __restrict__ K, uint16_t* __restrict__ Kb) {
    int t = blockIdx.x * blockDim.x + threadIdx.x;
    if (t >= S_LEN * KVHEADS * (HDIM / 2)) return;
    int j = t & 63;
    int h = (t >> 6) & (KVHEADS - 1);
    int s = t >> 8;
    float inv = __powf(10000.0f, -(float)(2 * j) * (1.0f / 128.0f));
    float sn, cs;
    __sincosf((float)s * inv, &sn, &cs);
    const float* src = K + (size_t)s * KVDIM + h * HDIM + 2 * j;
    float xr = src[0], xi = src[1];
    uint16_t* dst = Kb + ((size_t)h * S_LEN + s) * HDIM + 2 * j;
    dst[0] = f2bf(xr * cs - xi * sn);
    dst[1] = f2bf(xr * sn + xi * cs);
}

// V f32 [S][512] -> bf16 transposed [Hkv][128][S]  (d-major for B-frag loads)
__global__ void pack_vt(const float* __restrict__ V, uint16_t* __restrict__ Vt) {
    int t = blockIdx.x * blockDim.x + threadIdx.x;
    if (t >= S_LEN * KVDIM) return;
    int hd = t & (KVDIM - 1);
    int s  = t >> 9;
    int h  = hd >> 7;
    int d  = hd & (HDIM - 1);
    Vt[((size_t)h * HDIM + d) * S_LEN + s] = f2bf(V[(size_t)s * KVDIM + hd]);
}

// ---------------------------------------------------------------- attention
// One block per (128 query rows, head). 8 waves, each owns 16 query rows.
// Streams 32 keys/iter: QK^T via 8 WMMAs, online softmax in f32 regs,
// P -> LDS -> A-frag, PV via 8 WMMAs against transposed V.
__global__ __launch_bounds__(256, 1)
void attn_fa(const uint16_t* __restrict__ Qb, const uint16_t* __restrict__ Kb,
             const uint16_t* __restrict__ Vt, uint16_t* __restrict__ Ab) {
    __shared__ __align__(16) uint16_t sP[8][16 * 32];   // per-wave P tile

    const int tid  = threadIdx.x;
    const int lane = tid & 31;
    const int w    = tid >> 5;
    const int lrow = lane & 15;
    const int grp  = lane >> 4;
    const int h    = blockIdx.y;
    const int kvh  = h >> 2;                       // repeat_interleave(groups=4)
    const int q0   = blockIdx.x * 128 + w * 16;

    const uint16_t* Qh = Qb + (size_t)h   * S_LEN * HDIM;
    const uint16_t* Kh = Kb + (size_t)kvh * S_LEN * HDIM;
    const uint16_t* Vh = Vt + (size_t)kvh * HDIM  * S_LEN;

    bf16x16 qf[4];
    #pragma unroll
    for (int kk = 0; kk < 4; ++kk)
        qf[kk] = load_frag(Qh + (size_t)(q0 + lrow) * HDIM + kk * 32 + grp * 8);

    f32x8 o[8] = {};
    float mrow[8], lsum[8];
    #pragma unroll
    for (int r = 0; r < 8; ++r) { mrow[r] = -INFINITY; lsum[r] = 0.0f; }

    uint16_t* myP = sP[w];

    for (int kb = 0; kb < q0 + 16; kb += 32) {
        f32x8 s0 = {}, s1 = {};
        #pragma unroll
        for (int kk = 0; kk < 4; ++kk) {
            bf16x16 k0f = load_frag(Kh + (size_t)(kb      + lrow) * HDIM + kk * 32 + grp * 8);
            bf16x16 k1f = load_frag(Kh + (size_t)(kb + 16 + lrow) * HDIM + kk * 32 + grp * 8);
            s0 = wmma_bf16(qf[kk], k0f, s0);
            s1 = wmma_bf16(qf[kk], k1f, s1);
        }

        float alpha[8];
        #pragma unroll
        for (int r = 0; r < 8; ++r) {
            int row = q0 + r + grp * 8;
            float a = s0[r]; if (kb      + lrow > row) a = -INFINITY;
            float b = s1[r]; if (kb + 16 + lrow > row) b = -INFINITY;
            float cm = fmaxf(a, b);
            #pragma unroll
            for (int off = 1; off < 16; off <<= 1)
                cm = fmaxf(cm, __shfl_xor(cm, off, 32));   // within 16-lane half
            float mn = fmaxf(mrow[r], cm);
            float al = __expf(mrow[r] - mn);
            float p0 = __expf(a - mn);
            float p1 = __expf(b - mn);
            float rs = p0 + p1;
            #pragma unroll
            for (int off = 1; off < 16; off <<= 1)
                rs += __shfl_xor(rs, off, 32);
            lsum[r]  = lsum[r] * al + rs;
            mrow[r]  = mn;
            alpha[r] = al;
            myP[(r + grp * 8) * 32 + lrow]      = f2bf(p0);
            myP[(r + grp * 8) * 32 + 16 + lrow] = f2bf(p1);
        }
        #pragma unroll
        for (int nt = 0; nt < 8; ++nt)
            #pragma unroll
            for (int r = 0; r < 8; ++r)
                o[nt][r] *= alpha[r];

        asm volatile("s_wait_dscnt 0" ::: "memory");       // own-wave LDS RAW
        bf16x16 pf = load_frag(myP + lrow * 32 + grp * 8);

        #pragma unroll
        for (int nt = 0; nt < 8; ++nt) {
            bf16x16 vf = load_frag(Vh + (size_t)(nt * 16 + lrow) * S_LEN + kb + grp * 8);
            o[nt] = wmma_bf16(pf, vf, o[nt]);
        }
    }

    #pragma unroll
    for (int r = 0; r < 8; ++r) lsum[r] = 1.0f / lsum[r];
    #pragma unroll
    for (int nt = 0; nt < 8; ++nt)
        #pragma unroll
        for (int r = 0; r < 8; ++r) {
            int row = q0 + r + grp * 8;
            Ab[(size_t)row * DMODEL + h * HDIM + nt * 16 + lrow] =
                f2bf(o[nt][r] * lsum[r]);
        }
}

// ---------------------------------------------------------------- launch
extern "C" void kernel_launch(void* const* d_in, const int* in_sizes, int n_in,
                              void* d_out, int out_size, void* d_ws, size_t ws_size,
                              hipStream_t stream) {
    const float* x  = (const float*)d_in[0];
    const float* Wq = (const float*)d_in[1];
    const float* Wk = (const float*)d_in[2];
    const float* Wv = (const float*)d_in[3];
    const float* Wo = (const float*)d_in[4];
    float* out = (float*)d_out;

    char* p = (char*)d_ws;
    auto carve = [&](size_t bytes) {
        void* r = (void*)p;
        p += (bytes + 255) & ~(size_t)255;
        return r;
    };
    uint16_t* xb  = (uint16_t*)carve((size_t)S_LEN * DMODEL * 2);
    uint16_t* Wqb = (uint16_t*)carve((size_t)DMODEL * DMODEL * 2);
    uint16_t* Wkb = (uint16_t*)carve((size_t)KVDIM * DMODEL * 2);
    uint16_t* Wvb = (uint16_t*)carve((size_t)KVDIM * DMODEL * 2);
    uint16_t* Wob = (uint16_t*)carve((size_t)DMODEL * DMODEL * 2);
    float*    Qf  = (float*)   carve((size_t)S_LEN * DMODEL * 4);
    float*    Kf  = (float*)   carve((size_t)S_LEN * KVDIM * 4);
    float*    Vf  = (float*)   carve((size_t)S_LEN * KVDIM * 4);
    uint16_t* Qbf = (uint16_t*)carve((size_t)HEADS   * S_LEN * HDIM * 2);
    uint16_t* Kbf = (uint16_t*)carve((size_t)KVHEADS * S_LEN * HDIM * 2);
    uint16_t* Vtb = (uint16_t*)carve((size_t)KVHEADS * HDIM * S_LEN * 2);
    uint16_t* Ab  = (uint16_t*)carve((size_t)S_LEN * DMODEL * 2);

    auto cvt = [&](const float* src, uint16_t* dst, size_t n) {
        f32_to_bf16_kern<<<dim3((unsigned)((n + 255) / 256)), dim3(256), 0, stream>>>(
            src, dst, (int)n);
    };
    cvt(x,  xb,  (size_t)S_LEN * DMODEL);
    cvt(Wq, Wqb, (size_t)DMODEL * DMODEL);
    cvt(Wk, Wkb, (size_t)KVDIM * DMODEL);
    cvt(Wv, Wvb, (size_t)KVDIM * DMODEL);
    cvt(Wo, Wob, (size_t)DMODEL * DMODEL);

    // Projections: C = A * W^T
    gemm_bf16_nt<<<dim3(DMODEL / GBN, S_LEN / GBM), 256, 0, stream>>>(
        xb, Wqb, Qf, S_LEN, DMODEL, DMODEL);
    gemm_bf16_nt<<<dim3(KVDIM / GBN, S_LEN / GBM), 256, 0, stream>>>(
        xb, Wkb, Kf, S_LEN, KVDIM, DMODEL);
    gemm_bf16_nt<<<dim3(KVDIM / GBN, S_LEN / GBM), 256, 0, stream>>>(
        xb, Wvb, Vf, S_LEN, KVDIM, DMODEL);

    rope_pack_q<<<(S_LEN * HEADS   * (HDIM / 2)) / 256, 256, 0, stream>>>(Qf, Qbf);
    rope_pack_k<<<(S_LEN * KVHEADS * (HDIM / 2)) / 256, 256, 0, stream>>>(Kf, Kbf);
    pack_vt    <<<(S_LEN * KVDIM) / 256,               256, 0, stream>>>(Vf, Vtb);

    attn_fa<<<dim3(S_LEN / 128, HEADS), 256, 0, stream>>>(Qbf, Kbf, Vtb, Ab);

    // Output projection straight into d_out (f32)
    gemm_bf16_nt<<<dim3(DMODEL / GBN, S_LEN / GBM), 256, 0, stream>>>(
        Ab, Wob, out, S_LEN, DMODEL, DMODEL);
}
